// Net_6957847020044
// MI455X (gfx1250) — matrix operand order repaired and checked by reading, not compiled, not run on hardware
//
#include <hip/hip_runtime.h>
#include <stdint.h>

#define NSTEP 10
#define BATCH 16384
#define NIN   784
#define NH    300
#define NOUT  10

#define KPAD1 800    // 25 * 32
#define NPAD1 320    // 20 * 16
#define KPAD2 320    // 10 * 32
#define NPAD2 16

#define ROWS_PER_BLOCK 64
#define NT1 20       // N tiles in layer-1
#define KC1 25
#define KC2 10
#define RING 5       // B-fragment ring depth (within one kc chunk)

// LDS layout (bytes)
#define WROW_U16   40                               // 80 B padded row stride (32 elems + 16 B pad)
#define WBYTES     (NPAD1 * WROW_U16 * 2)           // 25,600 per W chunk buffer
#define MEM1_BYTES (ROWS_PER_BLOCK * NPAD1 * 4)     // 81,920
#define SPK1_BYTES (ROWS_PER_BLOCK * KPAD2 * 2)     // 40,960
#define WOFF       (MEM1_BYTES + SPK1_BYTES)        // 122,880
#define SMEM_TOTAL (WOFF + 2 * WBYTES)              // 174,080

#if defined(__has_builtin)
#  if __has_builtin(__builtin_amdgcn_tensor_load_to_lds) && __has_builtin(__builtin_amdgcn_s_wait_tensorcnt)
#    define USE_TDM 1
#  endif
#  if __has_builtin(__builtin_amdgcn_sched_group_barrier)
#    define SGB(mask, cnt, id) __builtin_amdgcn_sched_group_barrier((mask), (cnt), (id))
#  endif
#endif
#ifndef USE_TDM
#  define USE_TDM 0
#endif
#ifndef SGB
#  define SGB(mask, cnt, id)
#endif

typedef __attribute__((ext_vector_type(4)))  int           i4x;
typedef __attribute__((ext_vector_type(4)))  float         f4x;
typedef __attribute__((ext_vector_type(8)))  float         v8f;
typedef __attribute__((ext_vector_type(16))) __bf16        v16bf;
typedef __attribute__((ext_vector_type(4)))  unsigned int  u32x4;
typedef __attribute__((ext_vector_type(8)))  int           i32x8;
typedef __attribute__((ext_vector_type(4)))  int           i32x4;

union Frag {
    i4x      q[2];
    uint32_t u[8];
    v16bf    v;
};

struct RawA {
    f4x f0, f1, f2, f3;   // 16 fp32 = one lane's share of a 16x32 A tile
};

static __device__ __forceinline__ uint16_t f2bf(float f) {
    uint32_t u = __builtin_bit_cast(uint32_t, f);
    return (uint16_t)((u + 0x7FFFu + ((u >> 16) & 1u)) >> 16);  // RNE
}

static __device__ __forceinline__ uint32_t pack2(float lo, float hi) {
    uint32_t a = f2bf(lo);
    uint32_t b = f2bf(hi);
    return a | (b << 16);
}

static __device__ __forceinline__ Frag loadFrag16(const uint16_t* p) {
    Frag b;
    b.q[0] = *(const i4x*)(p);
    b.q[1] = *(const i4x*)(p + 16);
    return b;
}

static __device__ __forceinline__ RawA loadA(const float* xrow, int kc, int half) {
    const int k0 = kc * 32 + half * 8;
    f4x z = {0.0f, 0.0f, 0.0f, 0.0f};
    RawA r;
    r.f0 = z; r.f1 = z; r.f2 = z; r.f3 = z;
    if (k0 + 8 <= NIN)  { r.f0 = *(const f4x*)(xrow + k0);      r.f1 = *(const f4x*)(xrow + k0 + 4);  }
    if (k0 + 24 <= NIN) { r.f2 = *(const f4x*)(xrow + k0 + 16); r.f3 = *(const f4x*)(xrow + k0 + 20); }
    return r;
}

static __device__ __forceinline__ Frag cvtA(const RawA& r) {
    Frag a;
    a.u[0] = pack2(r.f0.x, r.f0.y); a.u[1] = pack2(r.f0.z, r.f0.w);
    a.u[2] = pack2(r.f1.x, r.f1.y); a.u[3] = pack2(r.f1.z, r.f1.w);
    a.u[4] = pack2(r.f2.x, r.f2.y); a.u[5] = pack2(r.f2.z, r.f2.w);
    a.u[6] = pack2(r.f3.x, r.f3.y); a.u[7] = pack2(r.f3.z, r.f3.w);
    return a;
}

#if USE_TDM
// DMA one W1 k-chunk [NPAD1 rows x 32 bf16] (row stride KPAD1) into LDS at lds_off,
// with 16B pad after each 64B row (LDS row stride 80B) for bank-conflict-free ds_load_b128.
static __device__ __forceinline__ void tdm_load_chunk(const uint16_t* gsrc, uint32_t lds_off) {
    const uint64_t ga = (uint64_t)(uintptr_t)gsrc;
    u32x4 g0 = (u32x4){
        1u,                                                   // count=1 (user descriptor)
        lds_off,                                              // lds_addr
        (uint32_t)ga,                                         // global_addr[31:0]
        (uint32_t)((ga >> 32) & 0x01FFFFFFu) | 0x80000000u    // global_addr[56:32] | type=2
    };
    // d0: data_size=1(2B)<<16 | pad_enable<<20 | pad_interval=3(16dw)<<22 | pad_amount=3(4dw)<<25
    i32x8 g1 = (i32x8){
        0x06D10000,            // flags (workgroup_mask=0)
        (int)(32u << 16),      // tensor_dim0 = 32 (low16 in [31:16]); atomic_barrier_addr=0
        (int)(320u << 16),     // tensor_dim0 hi=0; tensor_dim1 = 320 (low16)
        (int)(32u << 16),      // tensor_dim1 hi=0; tile_dim0 = 32
        320,                   // tile_dim1 = 320; tile_dim2 = 0
        KPAD1,                 // tensor_dim0_stride = 800 (elements)
        0, 0
    };
    i32x4 z4 = (i32x4){0, 0, 0, 0};
#if defined(__clang_major__) && (__clang_major__ >= 23)
    i32x8 z8 = (i32x8){0, 0, 0, 0, 0, 0, 0, 0};
    __builtin_amdgcn_tensor_load_to_lds(g0, g1, z4, z4, z8, 0);
#else
    __builtin_amdgcn_tensor_load_to_lds(g0, g1, z4, z4, 0);
#endif
}
#endif

// Fallback: cooperative synchronous staging of one chunk into LDS (same 80B row stride).
static __device__ __forceinline__ void stage_chunk_sync(const uint16_t* __restrict__ W1b,
                                                        uint16_t* __restrict__ wb,
                                                        int kc, int tid) {
    for (int j = tid; j < NPAD1 * 2; j += 128) {
        const int row = j >> 1, h = j & 1;
        const uint16_t* src = W1b + (size_t)row * KPAD1 + kc * 32 + h * 16;
        i4x a0 = *(const i4x*)src;
        i4x a1 = *(const i4x*)(src + 8);
        uint16_t* dst = wb + row * WROW_U16 + h * 16;
        *(i4x*)(dst)     = a0;
        *(i4x*)(dst + 8) = a1;
    }
}

// Convert W1 [300][784] f32 -> bf16 [320][800] zero-padded,
//         W2 [10][300]  f32 -> bf16 [16][320]  zero-padded.
__global__ void snn_prep(const float* __restrict__ W1, const float* __restrict__ W2,
                         uint16_t* __restrict__ W1b, uint16_t* __restrict__ W2b) {
    const int i = blockIdx.x * blockDim.x + threadIdx.x;
    const int n1 = NPAD1 * KPAD1;
    if (i < n1) {
        const int n = i / KPAD1, k = i % KPAD1;
        const float v = (n < NH && k < NIN) ? W1[n * NIN + k] : 0.0f;
        W1b[i] = f2bf(v);
    } else if (i < n1 + NPAD2 * KPAD2) {
        const int j = i - n1;
        const int n = j / KPAD2, k = j % KPAD2;
        const float v = (n < NOUT && k < NH) ? W2[n * NH + k] : 0.0f;
        W2b[j] = f2bf(v);
    }
}

__global__ __launch_bounds__(128) void snn_fused(
    const float* __restrict__ x,        // [NSTEP][BATCH][NIN] f32
    const uint16_t* __restrict__ W1b,   // [NPAD1][KPAD1] bf16 bits
    const uint16_t* __restrict__ W2b,   // [NPAD2][KPAD2] bf16 bits
    const float* __restrict__ b1p, const float* __restrict__ b2p,
    float* __restrict__ out)            // spk2[10][B][10] then mem2[10][B][10]
{
    extern __shared__ __align__(16) char smem[];
    float*    mem1s = (float*)smem;                      // [64][NPAD1]
    uint16_t* spk1s = (uint16_t*)(smem + MEM1_BYTES);    // [64][KPAD2]

    const int tid     = threadIdx.x;
    const int wave    = tid >> 5;
    const int lane    = tid & 31;
    const int lane16  = lane & 15;
    const int half    = lane >> 4;
    const int rowBase = wave * 16;
    const int gRowBase = blockIdx.x * ROWS_PER_BLOCK;

    const float beta1 = fminf(fmaxf(b1p[0], 0.0f), 1.0f);
    const float beta2 = fminf(fmaxf(b2p[0], 0.0f), 1.0f);

    // init persistent state
    for (int i = tid; i < ROWS_PER_BLOCK * NPAD1; i += 128) mem1s[i] = 0.0f;
    for (int i = tid; i < ROWS_PER_BLOCK * KPAD2; i += 128) spk1s[i] = 0;
    __syncthreads();

    float mem2[8];
    #pragma unroll
    for (int r = 0; r < 8; ++r) mem2[r] = 0.0f;

    const int aRow = gRowBase + rowBase + lane16;

#if USE_TDM
    const uint32_t ldsWbase = (uint32_t)(uintptr_t)smem + WOFF;   // LDS byte offset of W buffers
    // prologue: DMA chunk 0 into buffer 0
    if (wave == 0) {
        tdm_load_chunk(W1b, ldsWbase);
        __builtin_amdgcn_s_wait_tensorcnt(0);
    }
    __syncthreads();
#endif

    for (int t = 0; t < NSTEP; ++t) {
        const float* xrow = x + (size_t)t * BATCH * NIN + (size_t)aRow * NIN;

        // ---- layer 1 GEMM: cur1[64,320] = X[64,800] @ W1b^T (B staged in LDS) ----
        v8f acc[NT1];
        #pragma unroll
        for (int nt = 0; nt < NT1; ++nt) acc[nt] = (v8f){0,0,0,0,0,0,0,0};

        RawA rawc = loadA(xrow, 0, half);

        #pragma unroll 1
        for (int kc = 0; kc < KC1; ++kc) {
            const int par = (t + kc) & 1;
            uint16_t* wb = (uint16_t*)(smem + WOFF + (size_t)par * WBYTES);

#if USE_TDM
            // kick DMA for the next chunk (flattened across timesteps) into the other buffer
            if (wave == 0 && ((kc + 1 < KC1) || (t + 1 < NSTEP))) {
                const int nkc = (kc + 1 < KC1) ? (kc + 1) : 0;
                tdm_load_chunk(W1b + nkc * 32, ldsWbase + (uint32_t)((par ^ 1) * WBYTES));
            }
#else
            stage_chunk_sync(W1b, wb, kc, tid);
            __syncthreads();
#endif

            const Frag a = cvtA(rawc);
            if (kc + 1 < KC1) rawc = loadA(xrow, kc + 1, half);   // global x prefetch (divergent guards
                                                                  // kept OUT of the WMMA region below)

            // prime RING-deep B ring from LDS, then stream WMMAs
            const uint16_t* wl = wb + lane16 * WROW_U16 + half * 8;   // row n=lane16, K = 8h+s / 16+8h+s
            Frag bb[RING];
            #pragma unroll
            for (int j = 0; j < RING; ++j) bb[j] = loadFrag16(wl + j * 16 * WROW_U16);

            #pragma unroll
            for (int nt = 0; nt < NT1; ++nt) {
                acc[nt] = __builtin_amdgcn_wmma_f32_16x16x32_bf16(
                    false, a.v, false, bb[nt % RING].v, (short)0, acc[nt], false, false);
                const int pf = nt + RING;
                if (pf < NT1) bb[nt % RING] = loadFrag16(wl + pf * 16 * WROW_U16);
            }

            // Pin the pipeline shape: 2*RING DS-reads up front, then a steady
            // [1 WMMA, 2 DS-read] cadence (refills trail consumers by RING tiles),
            // then the RING-deep drain of WMMAs.
            SGB(0x100, 2 * RING, 0);                 // DS reads: priming fragments
            #pragma unroll
            for (int i = 0; i < NT1 - RING; ++i) {
                SGB(0x008, 1, 0);                    // 1 WMMA
                SGB(0x100, 2, 0);                    // 2 DS reads (refill)
            }
            SGB(0x008, RING, 0);                     // drain WMMAs

#if USE_TDM
            if (wave == 0) __builtin_amdgcn_s_wait_tensorcnt(0);  // next chunk landed
            __syncthreads();                                      // publish; also closes WAR window
#else
            __syncthreads();
#endif
        }

        // prefetch next timestep's x rows while we do LIF + layer 2
        if (t + 1 < NSTEP) {
            const float* xn = xrow + (size_t)BATCH * NIN;
            __builtin_prefetch(xn, 0, 3);
            __builtin_prefetch(xn + 64, 0, 3);
        }

        // ---- LIF 1 (reset-to-zero from previous mem), spikes -> LDS as bf16 ----
        #pragma unroll
        for (int nt = 0; nt < NT1; ++nt) {
            #pragma unroll
            for (int r = 0; r < 8; ++r) {
                const int m = rowBase + r + 8 * half;   // C layout: VGPR r -> M = r (+8 upper half)
                const int n = nt * 16 + lane16;
                const float cur  = acc[nt][r];
                const float mo   = mem1s[(size_t)m * NPAD1 + n];
                const float base = beta1 * mo + cur;
                const float mn   = (mo > 1.0f) ? 0.0f : base;
                mem1s[(size_t)m * NPAD1 + n] = mn;
                spk1s[(size_t)m * KPAD2 + n] = (mn > 1.0f) ? (uint16_t)0x3F80u : (uint16_t)0u;
            }
        }
        __syncthreads();

        // ---- layer 2 GEMM: cur2[64,16] = spk1[64,320] @ W2b^T ----
        v8f acc2 = (v8f){0,0,0,0,0,0,0,0};
        const uint16_t* s1row = spk1s + (size_t)(rowBase + lane16) * KPAD2;
        #pragma unroll
        for (int kc = 0; kc < KC2; ++kc) {
            const int k0 = kc * 32 + half * 8;
            Frag a2;
            a2.q[0] = *(const i4x*)(s1row + k0);
            a2.q[1] = *(const i4x*)(s1row + k0 + 16);
            const int kb = kc * 32 + half * 8;
            const uint16_t* p2 = W2b + (size_t)lane16 * KPAD2 + kb;
            Frag b2;
            b2.q[0] = *(const i4x*)(p2);
            b2.q[1] = *(const i4x*)(p2 + 16);
            acc2 = __builtin_amdgcn_wmma_f32_16x16x32_bf16(
                false, a2.v, false, b2.v, (short)0, acc2, false, false);
        }

        // ---- LIF 2 + store spk2/mem2 ----
        const size_t outSpkBase = (size_t)t * BATCH * NOUT;
        const size_t outMemBase = (size_t)NSTEP * BATCH * NOUT + outSpkBase;
        #pragma unroll
        for (int r = 0; r < 8; ++r) {
            const int m = rowBase + r + 8 * half;
            const int o = lane16;
            const float cur  = acc2[r];
            const float mo   = mem2[r];
            const float base = beta2 * mo + cur;
            const float mn   = (mo > 1.0f) ? 0.0f : base;
            mem2[r] = mn;
            if (o < NOUT) {
                const size_t gi = (size_t)(gRowBase + m) * NOUT + o;
                out[outSpkBase + gi] = (mn > 1.0f) ? 1.0f : 0.0f;
                out[outMemBase + gi] = mn;
            }
        }
        __syncthreads();  // protect spk1s before next timestep rewrites it
    }
}

extern "C" void kernel_launch(void* const* d_in, const int* in_sizes, int n_in,
                              void* d_out, int out_size, void* d_ws, size_t ws_size,
                              hipStream_t stream) {
    const float* x  = (const float*)d_in[0];
    const float* W1 = (const float*)d_in[1];
    const float* W2 = (const float*)d_in[2];
    const float* b1 = (const float*)d_in[3];
    const float* b2 = (const float*)d_in[4];

    uint16_t* W1b = (uint16_t*)d_ws;
    uint16_t* W2b = W1b + (size_t)NPAD1 * KPAD1;

    const int prepTotal = NPAD1 * KPAD1 + NPAD2 * KPAD2;
    snn_prep<<<(prepTotal + 255) / 256, 256, 0, stream>>>(W1, W2, W1b, W2b);

    snn_fused<<<BATCH / ROWS_PER_BLOCK, 128, SMEM_TOTAL, stream>>>(
        x, W1b, W2b, b1, b2, (float*)d_out);
}
